// PixelExchangeGuidedStateSpaceInteractionFusion_53687091200554
// MI455X (gfx1250) — compile-verified, matrix operand order checked by reading, not run on hardware
//
#include <hip/hip_runtime.h>

// ---------------------------------------------------------------------------
// CDNA5 (gfx1250, wave32) implementation of the PixelExchange / VSS / cross-
// attention fusion. All dense matmuls go through v_wmma_f32_16x16x32_bf16.
// Attention is flash-style (streaming softmax, no score materialization) and
// stages K/V tiles into LDS with GLOBAL_LOAD_ASYNC_TO_LDS_B128 (ASYNCcnt).
// The three selective scans run in one widened launch (24 workgroups).
// Workspace requirement: ~210 MB.
// ---------------------------------------------------------------------------

typedef __attribute__((ext_vector_type(16))) __bf16 v16bf;
typedef __attribute__((ext_vector_type(8)))  float  v8f;

constexpr int B_   = 2;
constexpr int H_   = 64;
constexpr int W_   = 64;
constexpr int L_   = 4096;     // H_*W_
constexpr int M_   = B_ * L_;  // 8192 rows for all pixel-major GEMMs

#define WMMA_BF16(a, b, c) \
  __builtin_amdgcn_wmma_f32_16x16x32_bf16(false, (a), false, (b), (short)0, (c), false, false)

// ---- CDNA5 async global->LDS copy (tracked by ASYNCcnt) --------------------
// lds_addr: per-lane LDS byte address (low 32 bits of the generic address of a
// __shared__ object == LDS offset per the flat-aperture mapping).
__device__ __forceinline__ void async_copy_b128(unsigned int lds_addr, const void* gaddr) {
  asm volatile("global_load_async_to_lds_b128 %0, %1, off"
               :
               : "v"(lds_addr), "v"((unsigned long long)(size_t)gaddr)
               : "memory");
}
__device__ __forceinline__ void wait_asynccnt0() {
  asm volatile("s_wait_asynccnt 0" ::: "memory");
}

// ---- WMMA fragment helpers (CDNA5 16-bit A layout / 32-bit C layout) -------
// 16-bit A (16xK): lanes 0-15 -> M=0..15 (K=0..7,16..23), lanes 16-31 mirror
// with K offset +8. Element e (0..15): vgpr j=e/2 holds packed pair.
// Each lane's 16 elements form two contiguous 8-element runs -> b128 loads.
__device__ __forceinline__ int frag_k(int e, int hi) {
  int j = e >> 1, s = e & 1;
  return (j < 4 ? 2 * j : 16 + 2 * (j - 4)) + s + hi * 8;
}

// A fragment: row-major A, p points at (row0, k0), ld = row stride.
__device__ __forceinline__ v16bf load_frag_a(const __bf16* p, int ld, int ln) {
  int m = ln & 15, hi = ln >> 4;
  v16bf r;
#pragma unroll
  for (int e = 0; e < 16; ++e) r[e] = p[(size_t)m * ld + frag_k(e, hi)];
  return r;
}

// B fragment from W stored (N,K) row-major: B[k][n] = W[n][k].
__device__ __forceinline__ v16bf load_frag_bt(const __bf16* p, int ld, int ln) {
  int n = ln & 15, hi = ln >> 4;
  v16bf r;
#pragma unroll
  for (int e = 0; e < 16; ++e) r[e] = p[(size_t)n * ld + frag_k(e, hi)];
  return r;
}

// B fragment from B stored (K,N) row-major. p points at (k-tile base, n0).
__device__ __forceinline__ v16bf load_frag_b(const __bf16* p, int ld, int ln) {
  int n = ln & 15, hi = ln >> 4;
  v16bf r;
#pragma unroll
  for (int e = 0; e < 16; ++e) r[e] = p[(size_t)frag_k(e, hi) * ld + n];
  return r;
}

// ---------------------------------------------------------------------------
// Generic GEMM:  out[M,N] = A[M,K] (bf16, row-major) x W[N,K]^T (bf16) + bias,
// optional +addsrc (f32, M x N), *oscale, dual f32 / bf16 stores.
// block = 128 (4 waves); wave w owns N-tile (blockIdx.y*4 + w).
// ---------------------------------------------------------------------------
__global__ __launch_bounds__(128) void gemm_bt_kernel(
    const __bf16* __restrict__ A, const __bf16* __restrict__ W,
    const float* __restrict__ bias, const float* __restrict__ addsrc,
    float* __restrict__ outF, __bf16* __restrict__ outB,
    int M, int N, int K, float oscale) {
  int wid = threadIdx.x >> 5;
  int ln  = threadIdx.x & 31;
  int m0  = blockIdx.x * 16;
  int n0  = (blockIdx.y * 4 + wid) * 16;
  if (n0 >= N || m0 >= M) return;

  v8f acc = {};
  for (int k0 = 0; k0 < K; k0 += 32) {
    v16bf a = load_frag_a(A + (size_t)m0 * K + k0, K, ln);
    v16bf b = load_frag_bt(W + (size_t)n0 * K + k0, K, ln);
    acc = WMMA_BF16(a, b, acc);
  }
  int n = ln & 15, hi = ln >> 4;
#pragma unroll
  for (int r = 0; r < 8; ++r) {
    int m = m0 + r + 8 * hi;
    int gn = n0 + n;
    size_t idx = (size_t)m * N + gn;
    float v = (acc[r] + bias[gn]) * oscale;
    if (addsrc) v += addsrc[idx];
    if (outF) outF[idx] = v;
    if (outB) outB[idx] = (__bf16)v;
  }
}

// ---------------------------------------------------------------------------
// Layout / precision conversion kernels
// ---------------------------------------------------------------------------
__global__ void f2b_kernel(const float* __restrict__ s, __bf16* __restrict__ d, int n) {
  int i = blockIdx.x * 256 + threadIdx.x;
  if (i < n) d[i] = (__bf16)s[i];
}

// (B,C,H,W) f32  ->  (B, L, C) bf16
__global__ void chw2lc_kernel(const float* __restrict__ src, __bf16* __restrict__ dst,
                              int C, int total) {
  int i = blockIdx.x * 256 + threadIdx.x;
  if (i >= total) return;
  int c = i % C;
  int l = (i / C) % L_;
  int b = i / (C * L_);
  dst[i] = (__bf16)src[((size_t)b * C + c) * L_ + l];
}

// ---------------------------------------------------------------------------
// Pixel exchange gather + LayerNorm(C=64):  ex[d] f32 (residual) + xln bf16.
// src stream s = (d + h%3 + w%3) % 3.   grid (B*L, 3), block 64.
// ---------------------------------------------------------------------------
__global__ __launch_bounds__(64) void exchange_ln_kernel(
    const float* __restrict__ feats, const float* __restrict__ ln_g,
    const float* __restrict__ ln_b, float* __restrict__ ex, __bf16* __restrict__ xln) {
  int bp = blockIdx.x, dg = blockIdx.y;
  int p = bp % L_, b = bp / L_;
  int hh = p / W_, ww = p % W_;
  int s = (dg + hh % 3 + ww % 3) % 3;
  int c = threadIdx.x;
  float v = feats[(((size_t)s * B_ + b) * L_ + p) * 64 + c];
  size_t oidx = (((size_t)dg * B_ + b) * L_ + p) * 64 + c;
  ex[oidx] = v;
  __shared__ float buf[64];
  buf[c] = v;
  __syncthreads();
  for (int st = 32; st > 0; st >>= 1) { if (c < st) buf[c] += buf[c + st]; __syncthreads(); }
  float mean = buf[0] / 64.f;
  __syncthreads();
  float dv = v - mean;
  buf[c] = dv * dv;
  __syncthreads();
  for (int st = 32; st > 0; st >>= 1) { if (c < st) buf[c] += buf[c + st]; __syncthreads(); }
  float var = buf[0] / 64.f;
  float o = dv * rsqrtf(var + 1e-5f) * ln_g[dg * 64 + c] + ln_b[dg * 64 + c];
  xln[oidx] = (__bf16)o;
}

// ---------------------------------------------------------------------------
// Depthwise 3x3 conv (SAME) + bias + SiLU on xc = xz[..., :128].
// xz pixel-major (B,L,256) f32 -> xcc (B,L,128) f32.  grid B*L, block 128.
// ---------------------------------------------------------------------------
__global__ __launch_bounds__(128) void dwconv_silu_kernel(
    const float* __restrict__ xz, const float* __restrict__ cw,
    const float* __restrict__ cb, float* __restrict__ xcc) {
  int bp = blockIdx.x;
  int p = bp % L_, b = bp / L_;
  int hh = p / W_, ww = p % W_;
  int d = threadIdx.x;
  float acc = cb[d];
#pragma unroll
  for (int kh = 0; kh < 3; ++kh)
#pragma unroll
    for (int kw = 0; kw < 3; ++kw) {
      int y = hh + kh - 1, x = ww + kw - 1;
      if (y >= 0 && y < H_ && x >= 0 && x < W_)
        acc += xz[((size_t)b * L_ + y * W_ + x) * 256 + d] * cw[d * 9 + kh * 3 + kw];
    }
  float s = acc / (1.f + __expf(-acc));
  xcc[((size_t)b * L_ + p) * 128 + d] = s;
}

// direction -> pixel mapping for the 4 scan orders
__device__ __forceinline__ int scan_pixel(int k, int l) {
  int lf = (k >= 2) ? (L_ - 1 - l) : l;
  if (k & 1) { int hh = lf % H_, ww = lf / H_; return hh * W_ + ww; }
  return lf;
}

// ---------------------------------------------------------------------------
// x_dbl = xproj @ xs,  dts = softplus(dt_w @ x_dbl[:4] + dt_b), Bs, Cs split.
// grid (L/32, B*4), block 128.  xs tile staged in LDS.
// ---------------------------------------------------------------------------
__global__ __launch_bounds__(128) void xdbl_kernel(
    const float* __restrict__ xcc, const float* __restrict__ xproj,
    const float* __restrict__ dtw, const float* __restrict__ dtb,
    float* __restrict__ Bss, float* __restrict__ Css, float* __restrict__ dts) {
  int bk = blockIdx.y, b = bk >> 2, k = bk & 3;
  int l0 = blockIdx.x * 32;
  __shared__ float xs[32][128];
  __shared__ float xr[32][4];
  for (int lt = 0; lt < 32; ++lt) {
    int p = scan_pixel(k, l0 + lt);
    xs[lt][threadIdx.x] = xcc[((size_t)b * L_ + p) * 128 + threadIdx.x];
  }
  __syncthreads();
  for (int t = threadIdx.x; t < 32 * 36; t += 128) {
    int lt = t / 36, c = t % 36;
    const float* wrow = xproj + (size_t)(k * 36 + c) * 128;
    float acc = 0.f;
#pragma unroll 8
    for (int dd = 0; dd < 128; ++dd) acc += xs[lt][dd] * wrow[dd];
    int l = l0 + lt;
    if (c < 4)       xr[lt][c] = acc;
    else if (c < 20) Bss[((size_t)bk * L_ + l) * 16 + (c - 4)]  = acc;
    else             Css[((size_t)bk * L_ + l) * 16 + (c - 20)] = acc;
  }
  __syncthreads();
  for (int t = threadIdx.x; t < 32 * 128; t += 128) {
    int lt = t >> 7, dd = t & 127;
    float acc = dtb[k * 128 + dd];
#pragma unroll
    for (int r = 0; r < 4; ++r) acc += xr[lt][r] * dtw[(size_t)(k * 128 + dd) * 4 + r];
    float sp = (acc > 20.f) ? acc : log1pf(__expf(acc));
    dts[((size_t)bk * L_ + l0 + lt) * 128 + dd] = sp;
  }
}

// ---------------------------------------------------------------------------
// Selective scan, all 3 streams in one launch: 4096 sequential steps, thread =
// (stream,b,k,d), 16 states in regs, B/C vectors staged via LDS in 64-step
// chunks.  grid 3*B*4 = 24 workgroups, block 128.
// ---------------------------------------------------------------------------
__global__ __launch_bounds__(128) void scan_kernel(
    const float* __restrict__ dts, const float* __restrict__ Bss,
    const float* __restrict__ Css, const float* __restrict__ xcc,
    const float* __restrict__ A_log, const float* __restrict__ Ds,
    float* __restrict__ ydir) {
  const size_t SS_D = (size_t)B_ * 4 * L_ * 128;  // dts / ydir per-stream size
  const size_t SS_B = (size_t)B_ * 4 * L_ * 16;   // Bs / Cs per-stream size
  const size_t SS_X = (size_t)B_ * L_ * 128;      // xcc per-stream size
  int i  = blockIdx.x / (B_ * 4);
  int bk = blockIdx.x % (B_ * 4);
  dts  += (size_t)i * SS_D;
  ydir += (size_t)i * SS_D;
  Bss  += (size_t)i * SS_B;
  Css  += (size_t)i * SS_B;
  xcc  += (size_t)i * SS_X;
  A_log += (size_t)i * 4 * 128 * 16;
  Ds    += (size_t)i * 4 * 128;

  int b = bk >> 2, k = bk & 3;
  int d = threadIdx.x;
  float Ac[16];
#pragma unroll
  for (int n = 0; n < 16; ++n) Ac[n] = -__expf(A_log[(size_t)(k * 128 + d) * 16 + n]);
  float Dv = Ds[k * 128 + d];
  float h[16];
#pragma unroll
  for (int n = 0; n < 16; ++n) h[n] = 0.f;
  __shared__ float Bsh[64][16], Csh[64][16];
  for (int l0 = 0; l0 < L_; l0 += 64) {
    __syncthreads();
    for (int t = threadIdx.x; t < 64 * 16; t += 128) {
      Bsh[t / 16][t % 16] = Bss[((size_t)bk * L_ + l0 + t / 16) * 16 + (t % 16)];
      Csh[t / 16][t % 16] = Css[((size_t)bk * L_ + l0 + t / 16) * 16 + (t % 16)];
    }
    __syncthreads();
    for (int li = 0; li < 64; ++li) {
      int l = l0 + li;
      int p = scan_pixel(k, l);
      float dt = dts[((size_t)bk * L_ + l) * 128 + d];
      float x  = xcc[((size_t)b * L_ + p) * 128 + d];
      float dx = dt * x;
      float y = 0.f;
#pragma unroll
      for (int n = 0; n < 16; ++n) {
        h[n] = __expf(dt * Ac[n]) * h[n] + dx * Bsh[li][n];
        y += h[n] * Csh[li][n];
      }
      ydir[((size_t)bk * L_ + l) * 128 + d] = y + Dv * x;
    }
  }
}

// ---------------------------------------------------------------------------
// Combine 4 scan directions at each pixel + LayerNorm(128) + SiLU(z) gate.
// grid B*L, block 128.
// ---------------------------------------------------------------------------
__global__ __launch_bounds__(128) void combine_ln_gate_kernel(
    const float* __restrict__ ydir, const float* __restrict__ xz,
    const float* __restrict__ og, const float* __restrict__ ob,
    __bf16* __restrict__ out) {
  int bp = blockIdx.x;
  int p = bp % L_, b = bp / L_;
  int hh = p / W_, ww = p % W_;
  int lwh = ww * H_ + hh;
  int d = threadIdx.x;
  float y = ydir[(((size_t)(b * 4 + 0)) * L_ + p) * 128 + d]
          + ydir[(((size_t)(b * 4 + 2)) * L_ + (L_ - 1 - p)) * 128 + d]
          + ydir[(((size_t)(b * 4 + 1)) * L_ + lwh) * 128 + d]
          + ydir[(((size_t)(b * 4 + 3)) * L_ + (L_ - 1 - lwh)) * 128 + d];
  __shared__ float buf[128];
  buf[d] = y;
  __syncthreads();
  for (int st = 64; st > 0; st >>= 1) { if (d < st) buf[d] += buf[d + st]; __syncthreads(); }
  float mean = buf[0] / 128.f;
  __syncthreads();
  float dv = y - mean;
  buf[d] = dv * dv;
  __syncthreads();
  for (int st = 64; st > 0; st >>= 1) { if (d < st) buf[d] += buf[d + st]; __syncthreads(); }
  float var = buf[0] / 128.f;
  float v = dv * rsqrtf(var + 1e-5f) * og[d] + ob[d];
  float z = xz[((size_t)b * L_ + p) * 256 + 128 + d];
  v *= z / (1.f + __expf(-z));
  out[((size_t)b * L_ + p) * 128 + d] = (__bf16)v;
}

// ---------------------------------------------------------------------------
// Flash attention: O = softmax(Q K^T) V  (scale folded into Q).
// block = 128 (4 waves); wave owns 16 query rows; K/V 32-key tiles staged into
// LDS with async global->LDS b128 copies; P re-packed C->A layout through
// per-wave LDS scratch.  grid (L/64, B).  accumulate: 0 = store, 1 = +=.
// ---------------------------------------------------------------------------
__global__ __launch_bounds__(128) void attention_kernel(
    const __bf16* __restrict__ Qb, const __bf16* __restrict__ Kb,
    const __bf16* __restrict__ Vb, float* __restrict__ outO, int accumulate) {
  int b = blockIdx.y;
  int wid = threadIdx.x >> 5;
  int ln  = threadIdx.x & 31;
  int q0  = blockIdx.x * 64 + wid * 16;

  __shared__ __bf16 Kt[32 * 64];
  __shared__ __bf16 Vt[32 * 64];
  __shared__ __bf16 Pt[4][16 * 32];

  // LDS byte offsets (low 32 bits of generic shared address = LDS offset)
  unsigned int kbase = (unsigned int)(size_t)(&Kt[0]);
  unsigned int vbase = (unsigned int)(size_t)(&Vt[0]);

  v16bf qa0 = load_frag_a(Qb + ((size_t)b * L_ + q0) * 64 + 0,  64, ln);
  v16bf qa1 = load_frag_a(Qb + ((size_t)b * L_ + q0) * 64 + 32, 64, ln);

  float rowm[8], rowl[8];
  v8f oacc[4] = {};
#pragma unroll
  for (int r = 0; r < 8; ++r) { rowm[r] = -1e30f; rowl[r] = 0.f; }

  for (int j0 = 0; j0 < L_; j0 += 32) {
    // Async-stage K/V 32x64 bf16 tiles into LDS (2 x b128 per thread each).
    const __bf16* Ks = Kb + ((size_t)b * L_ + j0) * 64;
    const __bf16* Vs = Vb + ((size_t)b * L_ + j0) * 64;
    for (int t = threadIdx.x; t < 256; t += 128) {
      async_copy_b128(kbase + t * 16, Ks + t * 8);
      async_copy_b128(vbase + t * 16, Vs + t * 8);
    }
    if (j0 + 32 < L_) {
      __builtin_prefetch(Kb + ((size_t)b * L_ + j0 + 32) * 64, 0, 1);
      __builtin_prefetch(Vb + ((size_t)b * L_ + j0 + 32) * 64, 0, 1);
    }
    wait_asynccnt0();
    __syncthreads();

    // S = Q @ K^T : 16 queries x 32 keys as two 16x16 C-tiles.
    v8f s0 = {}, s1 = {};
    {
      v16bf bt;
      bt = load_frag_bt(Kt + 0 * 64 + 0,  64, ln); s0 = WMMA_BF16(qa0, bt, s0);
      bt = load_frag_bt(Kt + 0 * 64 + 32, 64, ln); s0 = WMMA_BF16(qa1, bt, s0);
      bt = load_frag_bt(Kt + 16 * 64 + 0,  64, ln); s1 = WMMA_BF16(qa0, bt, s1);
      bt = load_frag_bt(Kt + 16 * 64 + 32, 64, ln); s1 = WMMA_BF16(qa1, bt, s1);
    }
    {
      int n = ln & 15, hi = ln >> 4;
#pragma unroll
      for (int r = 0; r < 8; ++r) {
        float smax = fmaxf(s0[r], s1[r]);
        for (int off = 1; off < 16; off <<= 1) smax = fmaxf(smax, __shfl_xor(smax, off));
        float mnew = fmaxf(rowm[r], smax);
        float alpha = __expf(rowm[r] - mnew);
        float p0 = __expf(s0[r] - mnew);
        float p1 = __expf(s1[r] - mnew);
        float ps = p0 + p1;
        for (int off = 1; off < 16; off <<= 1) ps += __shfl_xor(ps, off);
        rowl[r] = rowl[r] * alpha + ps;
        rowm[r] = mnew;
#pragma unroll
        for (int t = 0; t < 4; ++t) oacc[t][r] *= alpha;
        Pt[wid][(r + 8 * hi) * 32 + n]      = (__bf16)p0;
        Pt[wid][(r + 8 * hi) * 32 + 16 + n] = (__bf16)p1;
      }
    }
    __syncthreads();
    // O += P @ V  (P: 16x32 A-frag; V: 32x64 (K,N) tiles)
    v16bf pa = load_frag_a(&Pt[wid][0], 32, ln);
#pragma unroll
    for (int t = 0; t < 4; ++t) {
      v16bf vb = load_frag_b(Vt + t * 16, 64, ln);
      oacc[t] = WMMA_BF16(pa, vb, oacc[t]);
    }
    __syncthreads();
  }

  int n = ln & 15, hi = ln >> 4;
#pragma unroll
  for (int t = 0; t < 4; ++t)
#pragma unroll
    for (int r = 0; r < 8; ++r) {
      int m = q0 + r + 8 * hi;
      size_t idx = ((size_t)b * L_ + m) * 64 + t * 16 + n;
      float v = oacc[t][r] / rowl[r];
      if (accumulate) outO[idx] += v; else outO[idx] = v;
    }
}

// ---------------------------------------------------------------------------
// concat [fused(64) | v1(64) | v2(64)] -> bf16 (B,L,192)
// ---------------------------------------------------------------------------
__global__ void pack_cat_kernel(const float* __restrict__ fused,
                                const __bf16* __restrict__ v1b,
                                const __bf16* __restrict__ v2b,
                                __bf16* __restrict__ cat) {
  int i = blockIdx.x * 256 + threadIdx.x;
  if (i >= B_ * L_ * 192) return;
  int c = i % 192;
  size_t pl = (size_t)i / 192;
  __bf16 v;
  if (c < 64)       v = (__bf16)fused[pl * 64 + c];
  else if (c < 128) v = v1b[pl * 64 + (c - 64)];
  else              v = v2b[pl * 64 + (c - 128)];
  cat[i] = v;
}

// ---------------------------------------------------------------------------
// BatchNorm statistics per channel over (B, H, W), then normalize + ReLU
// writing (B, C, H, W) to d_out.
// ---------------------------------------------------------------------------
__global__ __launch_bounds__(256) void bn_stats_kernel(const float* __restrict__ x,
                                                       float* __restrict__ stats) {
  int c = blockIdx.x, t = threadIdx.x;
  float s = 0.f, ss = 0.f;
  for (int e = t; e < B_ * L_; e += 256) {
    float v = x[(size_t)e * 64 + c];
    s += v; ss += v * v;
  }
  __shared__ float b1[256], b2[256];
  b1[t] = s; b2[t] = ss;
  __syncthreads();
  for (int st = 128; st > 0; st >>= 1) {
    if (t < st) { b1[t] += b1[t + st]; b2[t] += b2[t + st]; }
    __syncthreads();
  }
  if (t == 0) {
    float m = b1[0] / (float)(B_ * L_);
    stats[c] = m;
    stats[64 + c] = b2[0] / (float)(B_ * L_) - m * m;
  }
}

__global__ void bn_relu_kernel(const float* __restrict__ x, const float* __restrict__ stats,
                               const float* __restrict__ g, const float* __restrict__ bb,
                               float* __restrict__ out) {
  int i = blockIdx.x * 256 + threadIdx.x;
  if (i >= B_ * 64 * L_) return;
  int p = i % L_;
  int c = (i / L_) % 64;
  int b = i / (64 * L_);
  float v = (x[((size_t)b * L_ + p) * 64 + c] - stats[c]) * rsqrtf(stats[64 + c] + 1e-5f) * g[c] + bb[c];
  out[i] = v > 0.f ? v : 0.f;
}

// ---------------------------------------------------------------------------
// Host side
// ---------------------------------------------------------------------------
static void* wsAlloc(char* base, size_t& cur, size_t bytes) {
  void* p = base + cur;
  cur += (bytes + 255) & ~(size_t)255;
  return p;
}

extern "C" void kernel_launch(void* const* d_in, const int* in_sizes, int n_in,
                              void* d_out, int out_size, void* d_ws, size_t ws_size,
                              hipStream_t stream) {
  (void)in_sizes; (void)n_in; (void)out_size; (void)ws_size;
  const float* x1      = (const float*)d_in[0];
  const float* x2      = (const float*)d_in[1];
  const float* x3      = (const float*)d_in[2];
  const float* aw1     = (const float*)d_in[3];
  const float* ab1     = (const float*)d_in[4];
  const float* aw2     = (const float*)d_in[5];
  const float* ab2     = (const float*)d_in[6];
  const float* aw3     = (const float*)d_in[7];
  const float* ab3     = (const float*)d_in[8];
  const float* ln_g    = (const float*)d_in[9];
  const float* ln_b    = (const float*)d_in[10];
  const float* in_w    = (const float*)d_in[11];
  const float* in_b    = (const float*)d_in[12];
  const float* conv_w  = (const float*)d_in[13];
  const float* conv_b  = (const float*)d_in[14];
  const float* xproj_w = (const float*)d_in[15];
  const float* dt_w    = (const float*)d_in[16];
  const float* dt_b    = (const float*)d_in[17];
  const float* A_log   = (const float*)d_in[18];
  const float* Ds      = (const float*)d_in[19];
  const float* onorm_g = (const float*)d_in[20];
  const float* onorm_b = (const float*)d_in[21];
  const float* out_w   = (const float*)d_in[22];
  const float* out_b   = (const float*)d_in[23];
  const float* k1_w    = (const float*)d_in[24];
  const float* k1_b    = (const float*)d_in[25];
  const float* k2_w    = (const float*)d_in[26];
  const float* k2_b    = (const float*)d_in[27];
  const float* q3_w    = (const float*)d_in[28];
  const float* q3_b    = (const float*)d_in[29];
  const float* v3_w    = (const float*)d_in[30];
  const float* v3_b    = (const float*)d_in[31];
  const float* proj_w  = (const float*)d_in[32];
  const float* proj_b  = (const float*)d_in[33];
  const float* bn_g    = (const float*)d_in[34];
  const float* bn_b    = (const float*)d_in[35];
  float* outp = (float*)d_out;

  char* ws = (char*)d_ws;
  size_t cur = 0;
  const size_t BL = (size_t)B_ * L_;  // 8192

  __bf16* x1b    = (__bf16*)wsAlloc(ws, cur, BL * 128 * 2);
  __bf16* x2b    = (__bf16*)wsAlloc(ws, cur, BL * 256 * 2);
  __bf16* x3b    = (__bf16*)wsAlloc(ws, cur, BL * 512 * 2);
  __bf16* aw1b   = (__bf16*)wsAlloc(ws, cur, 64 * 128 * 2);
  __bf16* aw2b   = (__bf16*)wsAlloc(ws, cur, 64 * 256 * 2);
  __bf16* aw3b   = (__bf16*)wsAlloc(ws, cur, 64 * 512 * 2);
  __bf16* inwb   = (__bf16*)wsAlloc(ws, cur, 3 * 256 * 64 * 2);
  __bf16* outwb  = (__bf16*)wsAlloc(ws, cur, 3 * 64 * 128 * 2);
  __bf16* k1wb   = (__bf16*)wsAlloc(ws, cur, 64 * 64 * 2);
  __bf16* k2wb   = (__bf16*)wsAlloc(ws, cur, 64 * 64 * 2);
  __bf16* q3wb   = (__bf16*)wsAlloc(ws, cur, 64 * 64 * 2);
  __bf16* v3wb   = (__bf16*)wsAlloc(ws, cur, 64 * 64 * 2);
  __bf16* projwb = (__bf16*)wsAlloc(ws, cur, 64 * 192 * 2);
  float*  feats  = (float*)wsAlloc(ws, cur, 3 * BL * 64 * 4);
  float*  ex     = (float*)wsAlloc(ws, cur, 3 * BL * 64 * 4);
  __bf16* xln    = (__bf16*)wsAlloc(ws, cur, 3 * BL * 64 * 2);
  float*  xz3    = (float*)wsAlloc(ws, cur, 3 * BL * 256 * 4);
  float*  xcc3   = (float*)wsAlloc(ws, cur, 3 * BL * 128 * 4);
  float*  Bss3   = (float*)wsAlloc(ws, cur, 3 * BL * 4 * 16 * 4);
  float*  Css3   = (float*)wsAlloc(ws, cur, 3 * BL * 4 * 16 * 4);
  float*  dts3   = (float*)wsAlloc(ws, cur, 3 * BL * 4 * 128 * 4);
  float*  ydir3  = (float*)wsAlloc(ws, cur, 3 * BL * 4 * 128 * 4);
  __bf16* ycomb  = (__bf16*)wsAlloc(ws, cur, BL * 128 * 2);
  float*  vss    = (float*)wsAlloc(ws, cur, 3 * BL * 64 * 4);
  __bf16* vssB   = (__bf16*)wsAlloc(ws, cur, 3 * BL * 64 * 2);
  __bf16* Qb     = (__bf16*)wsAlloc(ws, cur, BL * 64 * 2);
  __bf16* K1b    = (__bf16*)wsAlloc(ws, cur, BL * 64 * 2);
  __bf16* K2b    = (__bf16*)wsAlloc(ws, cur, BL * 64 * 2);
  __bf16* Vb     = (__bf16*)wsAlloc(ws, cur, BL * 64 * 2);
  float*  fused  = (float*)wsAlloc(ws, cur, BL * 64 * 4);
  __bf16* catB   = (__bf16*)wsAlloc(ws, cur, BL * 192 * 2);
  float*  projo  = (float*)wsAlloc(ws, cur, BL * 64 * 4);
  float*  stats  = (float*)wsAlloc(ws, cur, 128 * 4);

  auto cdiv = [](int a, int b) { return (a + b - 1) / b; };

  // --- input transposes + weight conversions to bf16 -----------------------
  chw2lc_kernel<<<cdiv((int)(BL * 128), 256), 256, 0, stream>>>(x1, x1b, 128, (int)(BL * 128));
  chw2lc_kernel<<<cdiv((int)(BL * 256), 256), 256, 0, stream>>>(x2, x2b, 256, (int)(BL * 256));
  chw2lc_kernel<<<cdiv((int)(BL * 512), 256), 256, 0, stream>>>(x3, x3b, 512, (int)(BL * 512));
  f2b_kernel<<<cdiv(64 * 128, 256), 256, 0, stream>>>(aw1, aw1b, 64 * 128);
  f2b_kernel<<<cdiv(64 * 256, 256), 256, 0, stream>>>(aw2, aw2b, 64 * 256);
  f2b_kernel<<<cdiv(64 * 512, 256), 256, 0, stream>>>(aw3, aw3b, 64 * 512);
  f2b_kernel<<<cdiv(3 * 256 * 64, 256), 256, 0, stream>>>(in_w, inwb, 3 * 256 * 64);
  f2b_kernel<<<cdiv(3 * 64 * 128, 256), 256, 0, stream>>>(out_w, outwb, 3 * 64 * 128);
  f2b_kernel<<<cdiv(64 * 64, 256), 256, 0, stream>>>(k1_w, k1wb, 64 * 64);
  f2b_kernel<<<cdiv(64 * 64, 256), 256, 0, stream>>>(k2_w, k2wb, 64 * 64);
  f2b_kernel<<<cdiv(64 * 64, 256), 256, 0, stream>>>(q3_w, q3wb, 64 * 64);
  f2b_kernel<<<cdiv(64 * 64, 256), 256, 0, stream>>>(v3_w, v3wb, 64 * 64);
  f2b_kernel<<<cdiv(64 * 192, 256), 256, 0, stream>>>(proj_w, projwb, 64 * 192);

  // --- channel-align conv1x1s (WMMA GEMMs) ---------------------------------
  gemm_bt_kernel<<<dim3(M_ / 16, 1), 128, 0, stream>>>(x1b, aw1b, ab1, nullptr,
      feats + 0 * BL * 64, nullptr, M_, 64, 128, 1.f);
  gemm_bt_kernel<<<dim3(M_ / 16, 1), 128, 0, stream>>>(x2b, aw2b, ab2, nullptr,
      feats + 1 * BL * 64, nullptr, M_, 64, 256, 1.f);
  gemm_bt_kernel<<<dim3(M_ / 16, 1), 128, 0, stream>>>(x3b, aw3b, ab3, nullptr,
      feats + 2 * BL * 64, nullptr, M_, 64, 512, 1.f);

  // --- pixel exchange + LN -------------------------------------------------
  exchange_ln_kernel<<<dim3((int)BL, 3), 64, 0, stream>>>(feats, ln_g, ln_b, ex, xln);

  // --- VSS phase A: pre-scan per stream ------------------------------------
  for (int i = 0; i < 3; ++i) {
    gemm_bt_kernel<<<dim3(M_ / 16, 4), 128, 0, stream>>>(
        xln + (size_t)i * BL * 64, inwb + (size_t)i * 256 * 64, in_b + i * 256,
        nullptr, xz3 + (size_t)i * BL * 256, nullptr, M_, 256, 64, 1.f);
    dwconv_silu_kernel<<<(int)BL, 128, 0, stream>>>(
        xz3 + (size_t)i * BL * 256, conv_w + (size_t)i * 128 * 9, conv_b + i * 128,
        xcc3 + (size_t)i * BL * 128);
    xdbl_kernel<<<dim3(L_ / 32, B_ * 4), 128, 0, stream>>>(
        xcc3 + (size_t)i * BL * 128, xproj_w + (size_t)i * 4 * 36 * 128,
        dt_w + (size_t)i * 4 * 128 * 4, dt_b + i * 4 * 128,
        Bss3 + (size_t)i * BL * 4 * 16, Css3 + (size_t)i * BL * 4 * 16,
        dts3 + (size_t)i * BL * 4 * 128);
  }

  // --- VSS phase B: all 3 streams' scans in one widened launch -------------
  scan_kernel<<<3 * B_ * 4, 128, 0, stream>>>(dts3, Bss3, Css3, xcc3, A_log, Ds, ydir3);

  // --- VSS phase C: combine + gate + out_proj (+ residual) per stream ------
  for (int i = 0; i < 3; ++i) {
    combine_ln_gate_kernel<<<(int)BL, 128, 0, stream>>>(
        ydir3 + (size_t)i * BL * 4 * 128, xz3 + (size_t)i * BL * 256,
        onorm_g + i * 128, onorm_b + i * 128, ycomb);
    gemm_bt_kernel<<<dim3(M_ / 16, 1), 128, 0, stream>>>(
        ycomb, outwb + (size_t)i * 64 * 128, out_b + i * 64, ex + (size_t)i * BL * 64,
        vss + (size_t)i * BL * 64, vssB + (size_t)i * BL * 64, M_, 64, 128, 1.f);
  }

  // --- Q / K1 / K2 / V GEMMs (softmax scale 1/sqrt(64) folded into Q) ------
  gemm_bt_kernel<<<dim3(M_ / 16, 1), 128, 0, stream>>>(vssB + 0 * BL * 64, k1wb, k1_b,
      nullptr, nullptr, K1b, M_, 64, 64, 1.f);
  gemm_bt_kernel<<<dim3(M_ / 16, 1), 128, 0, stream>>>(vssB + 1 * BL * 64, k2wb, k2_b,
      nullptr, nullptr, K2b, M_, 64, 64, 1.f);
  gemm_bt_kernel<<<dim3(M_ / 16, 1), 128, 0, stream>>>(vssB + 2 * BL * 64, q3wb, q3_b,
      nullptr, nullptr, Qb, M_, 64, 64, 0.125f);
  gemm_bt_kernel<<<dim3(M_ / 16, 1), 128, 0, stream>>>(vssB + 2 * BL * 64, v3wb, v3_b,
      nullptr, nullptr, Vb, M_, 64, 64, 1.f);

  // --- two flash attentions accumulating into `fused` ----------------------
  attention_kernel<<<dim3(L_ / 64, B_), 128, 0, stream>>>(Qb, K1b, Vb, fused, 0);
  attention_kernel<<<dim3(L_ / 64, B_), 128, 0, stream>>>(Qb, K2b, Vb, fused, 1);

  // --- concat + output projection + BN + ReLU ------------------------------
  pack_cat_kernel<<<cdiv((int)(BL * 192), 256), 256, 0, stream>>>(
      fused, vssB + 0 * BL * 64, vssB + 1 * BL * 64, catB);
  gemm_bt_kernel<<<dim3(M_ / 16, 1), 128, 0, stream>>>(catB, projwb, proj_b,
      nullptr, projo, nullptr, M_, 64, 192, 1.f);
  bn_stats_kernel<<<64, 256, 0, stream>>>(projo, stats);
  bn_relu_kernel<<<cdiv((int)(BL * 64), 256), 256, 0, stream>>>(projo, stats, bn_g, bn_b, outp);
}
